// GreenTF_34505767256674
// MI455X (gfx1250) — matrix operand order repaired and checked by reading, not compiled, not run on hardware
//
#include <hip/hip_runtime.h>
#include <math.h>

typedef __attribute__((ext_vector_type(16))) _Float16 v16h;
typedef __attribute__((ext_vector_type(8)))  _Float16 v8h;
typedef __attribute__((ext_vector_type(8)))  float    v8f;

// ---- problem constants (from the reference) ----
#define KSIZE_   2048      // FIR window length
#define NF2_     16002     // 2 * NUM_FREQS rows of the kernel matrix
#define NFREQ_   8001
#define MPAD_    16128     // 126 * 128 : M padded for the GEMM
#define B_       2
#define TWAV_    32000
#define TOUT_    2000
#define TPAD_    2048      // N padded to 32 * 64
#define STRIDE_  16

// ---- GEMM tiling ----
#define BM  128
#define BN  64
#define BK  64
#define LDA 72             // LDS row stride in f16 (+8 pad, keeps 16B alignment)
#define LDB 72
#define NSTAGES (KSIZE_ / BK)   // 32

// -------------------------------------------------------------------------
// Kernel 0: multiplier = sqrt(8) / sum(envelope)   (the two /SR cancel)
// -------------------------------------------------------------------------
__global__ void env_mult_kernel(const float* __restrict__ env, float* __restrict__ mult) {
    __shared__ float red[256];
    float s = 0.0f;
    for (int i = threadIdx.x; i < KSIZE_; i += 256) s += env[i];
    red[threadIdx.x] = s;
    __syncthreads();
    for (int off = 128; off > 0; off >>= 1) {
        if ((int)threadIdx.x < off) red[threadIdx.x] += red[threadIdx.x + off];
        __syncthreads();
    }
    if (threadIdx.x == 0) mult[0] = sqrtf(8.0f) / red[0];
}

// -------------------------------------------------------------------------
// Kernel 1: A16[f][k] = f16( kernel[f][k] * envelope[k] ), zero-padded rows
// -------------------------------------------------------------------------
__global__ void a_prep(const float* __restrict__ kern, const float* __restrict__ env,
                       _Float16* __restrict__ A16) {
    int i = blockIdx.x * 256 + threadIdx.x;       // < MPAD_*KSIZE_ = 33,030,144
    int f = i >> 11;
    int k = i & (KSIZE_ - 1);
    float v = 0.0f;
    if (f < NF2_) v = kern[f * KSIZE_ + k] * env[k];
    A16[i] = (_Float16)v;
}

// -------------------------------------------------------------------------
// Kernel 2: F16[b][t][k] = f16( wav[b][t*16 + k - 2047] ), zero outside
// -------------------------------------------------------------------------
__global__ void b_prep(const float* __restrict__ wav, _Float16* __restrict__ F16) {
    int i = blockIdx.x * 256 + threadIdx.x;       // < B_*TPAD_*KSIZE_ = 8,388,608
    int b = i >> 22;
    int r = i & ((1 << 22) - 1);
    int t = r >> 11;
    int k = r & (KSIZE_ - 1);
    int src = t * STRIDE_ + k - (KSIZE_ - 1);
    float v = 0.0f;
    if (t < TOUT_ && src >= 0 && src < TWAV_) v = wav[b * TWAV_ + src];
    F16[i] = (_Float16)v;
}

// -------------------------------------------------------------------------
// Async global -> LDS staging helpers (CDNA5 GLOBAL_LOAD_ASYNC_TO_LDS_B128,
// tracked by ASYNCcnt). saddr form: 64-bit SGPR base + 32-bit VGPR byte off.
// Each call moves 32 lanes x 16B = 512B directly into LDS.
// -------------------------------------------------------------------------
__device__ __forceinline__ void async_b128(unsigned lds_byte_addr,
                                           unsigned gbyte_off,
                                           const _Float16* gbase) {
    asm volatile("global_load_async_to_lds_b128 %0, %1, %2"
                 :: "v"(lds_byte_addr), "v"(gbyte_off), "s"(gbase)
                 : "memory");
}

__device__ __forceinline__ unsigned lds_addr32(const void* p) {
    // shared-aperture flat address: low 32 bits == group-segment byte offset
    return (unsigned)(unsigned long long)p;
}

// Issue one stage (6 async ops per wave): A panel BM x BK, B panel BN x BK
__device__ __forceinline__ void issue_stage(const _Float16* Abase, const _Float16* Bbase,
                                            unsigned ldsA, unsigned ldsB,
                                            int m0, int n0, int kc,
                                            int wid, int lane) {
    #pragma unroll
    for (int it = 0; it < 4; ++it) {                 // 32 wave-ops cover 16KB
        int q  = (it * 8 + wid) * 32 + lane;         // 0..1023
        int r  = q >> 3;                             // 8 chunks per row
        int c8 = (q & 7) * 8;
        async_b128(ldsA + (unsigned)(r * LDA + c8) * 2,
                   (unsigned)(((m0 + r) * KSIZE_ + kc + c8) * 2), Abase);
    }
    #pragma unroll
    for (int it = 0; it < 2; ++it) {                 // 16 wave-ops cover 8KB
        int q  = (it * 8 + wid) * 32 + lane;         // 0..511
        int r  = q >> 3;
        int c8 = (q & 7) * 8;
        async_b128(ldsB + (unsigned)(r * LDB + c8) * 2,
                   (unsigned)(((n0 + r) * KSIZE_ + kc + c8) * 2), Bbase);
    }
}

// -------------------------------------------------------------------------
// Kernel 3: C[b] = A16 (16128x2048) x F16[b]^T, f16 WMMA, f32 acc.
// 8 waves per block (4Mx2N), 2x2 WMMA tiles per wave, async double-buffered.
// -------------------------------------------------------------------------
__global__ void __launch_bounds__(256)
gemm_wmma(const _Float16* __restrict__ A16, const _Float16* __restrict__ F16,
          const float* __restrict__ multp, float* __restrict__ out) {
    __shared__ __align__(16) _Float16 Ash[2][BM * LDA];
    __shared__ __align__(16) _Float16 Bsh[2][BN * LDB];

    const int tid   = threadIdx.x;
    const int lane  = tid & 31;
    const int wid   = tid >> 5;      // 0..7
    const int wm    = wid & 3;       // M: wm*32
    const int wn    = wid >> 2;      // N: wn*32
    const int mlane = lane & 15;
    const int hi    = lane >> 4;     // half-wave select

    const int m0 = blockIdx.x * BM;
    const int n0 = blockIdx.y * BN;
    const int bb = blockIdx.z;

    const _Float16* __restrict__ Bbase = F16 + (size_t)bb * TPAD_ * KSIZE_;

    const unsigned ldsA[2] = { lds_addr32(&Ash[0][0]), lds_addr32(&Ash[1][0]) };
    const unsigned ldsB[2] = { lds_addr32(&Bsh[0][0]), lds_addr32(&Bsh[1][0]) };

    v8f acc[2][2] = {};

    // prologue: fill buffer 0
    issue_stage(A16, Bbase, ldsA[0], ldsB[0], m0, n0, 0, wid, lane);

    for (int ks = 0; ks < NSTAGES; ++ks) {
        const int cur = ks & 1;
        if (ks + 1 < NSTAGES) {
            issue_stage(A16, Bbase, ldsA[cur ^ 1], ldsB[cur ^ 1],
                        m0, n0, (ks + 1) * BK, wid, lane);
            // 6 ops just issued for stage ks+1 outstanding => stage ks landed
            asm volatile("s_wait_asynccnt 0x6" ::: "memory");
        } else {
            asm volatile("s_wait_asynccnt 0x0" ::: "memory");
        }
        __syncthreads();

        const _Float16* Abuf = &Ash[cur][0];
        const _Float16* Bbuf = &Bsh[cur][0];

        #pragma unroll
        for (int ko = 0; ko < BK; ko += 32) {
            // B fragments (32x16, KxN): lane = column n, K = {ko+hi*8..+7, ko+16+hi*8..+7}
            v16h bfrag[2];
            #pragma unroll
            for (int tn = 0; tn < 2; ++tn) {
                const _Float16* p = Bbuf + (wn * 32 + tn * 16 + mlane) * LDB + ko + hi * 8;
                v8h lo  = *(const v8h*)p;
                v8h hi8 = *(const v8h*)(p + 16);
                bfrag[tn] = __builtin_shufflevector(lo, hi8,
                    0,1,2,3,4,5,6,7,8,9,10,11,12,13,14,15);
            }
            #pragma unroll
            for (int tm = 0; tm < 2; ++tm) {
                // A fragment (16x32): lane = row m, K = {ko+hi*8..+7, ko+16+hi*8..+7}
                const _Float16* p = Abuf + (wm * 32 + tm * 16 + mlane) * LDA + ko + hi * 8;
                v8h lo  = *(const v8h*)p;
                v8h hi8 = *(const v8h*)(p + 16);
                v16h afrag = __builtin_shufflevector(lo, hi8,
                    0,1,2,3,4,5,6,7,8,9,10,11,12,13,14,15);
                #pragma unroll
                for (int tn = 0; tn < 2; ++tn) {
                    acc[tm][tn] = __builtin_amdgcn_wmma_f32_16x16x32_f16(
                        false, afrag, false, bfrag[tn],
                        (short)0, acc[tm][tn], false, false);
                }
            }
        }
        __syncthreads();   // buffer `cur` free for stage ks+2
    }

    // ---- store: C VGPR r holds (M = r + hi*8, N = mlane); route sin/cos ----
    const float mult = multp[0];
    #pragma unroll
    for (int tm = 0; tm < 2; ++tm) {
        #pragma unroll
        for (int tn = 0; tn < 2; ++tn) {
            int t = n0 + wn * 32 + tn * 16 + mlane;
            if (t >= TOUT_) continue;
            #pragma unroll
            for (int r = 0; r < 8; ++r) {
                int f = m0 + wm * 32 + tm * 16 + r + hi * 8;
                if (f >= NF2_) continue;
                float v = acc[tm][tn][r] * mult;
                size_t o;
                if (f < NFREQ_)
                    o = ((size_t)bb * NFREQ_ + f) * TOUT_ + t;
                else
                    o = (size_t)B_ * NFREQ_ * TOUT_
                      + ((size_t)bb * NFREQ_ + (f - NFREQ_)) * TOUT_ + t;
                out[o] = v;
            }
        }
    }
}

// -------------------------------------------------------------------------
extern "C" void kernel_launch(void* const* d_in, const int* in_sizes, int n_in,
                              void* d_out, int out_size, void* d_ws, size_t ws_size,
                              hipStream_t stream) {
    const float* wav  = (const float*)d_in[0];   // (2, 32000) f32
    const float* kern = (const float*)d_in[1];   // (16002, 2048) f32
    const float* env  = (const float*)d_in[2];   // (2048,) f32
    float* out = (float*)d_out;                  // sspec ++ cspec, f32

    // workspace layout: [0,256) scalar multiplier | A16 (66.06 MB) | F16 (16.78 MB)
    char* ws = (char*)d_ws;
    float*    mult = (float*)ws;
    _Float16* A16  = (_Float16*)(ws + 256);
    _Float16* F16  = (_Float16*)(ws + 256 + (size_t)MPAD_ * KSIZE_ * sizeof(_Float16));

    env_mult_kernel<<<1, 256, 0, stream>>>(env, mult);
    a_prep<<<(MPAD_ * KSIZE_) / 256, 256, 0, stream>>>(kern, env, A16);
    b_prep<<<(B_ * TPAD_ * KSIZE_) / 256, 256, 0, stream>>>(wav, F16);

    dim3 grid(MPAD_ / BM, TPAD_ / BN, B_);
    gemm_wmma<<<grid, 256, 0, stream>>>(A16, F16, mult, out);
}
